// SSMBlock_60138132079076
// MI455X (gfx1250) — matrix operand order repaired and checked by reading, not compiled, not run on hardware
//
#include <hip/hip_runtime.h>

// ---------------------------------------------------------------------------
// SSM block for MI455X (gfx1250, wave32, WMMA 16x16x32 bf16).
// Phase 1: bx = x @ b_mat  (bf16 WMMA GEMM, f32 accumulate, LDS-staged A with
//          async-to-LDS double buffering) -> d_out in place.
// Phase 2: persistent scan kernel, 8 WGs, device-wide atomic barrier per step,
//          h kept bf16 in L2, double-buffered.
// ---------------------------------------------------------------------------

typedef __attribute__((ext_vector_type(16))) __bf16 v16bf;
typedef __attribute__((ext_vector_type(8)))  float  v8f;
typedef __attribute__((ext_vector_type(4)))  unsigned int u32x4;
typedef __attribute__((ext_vector_type(4)))  int v4i;

#define H 2048
#define T 2048
#define BB 8
#define KT 64            // K tiles of 32
#define NT 128           // N tiles of 16
#define NWG 8            // persistent workgroups in scan
#define HBUF 32768       // one h buffer: 16*2048 bf16 elements

// ---- async-to-LDS support (guarded; falls back to ds_store staging) ---------
#if __has_builtin(__builtin_amdgcn_global_load_async_to_lds_b128)
  #define HAVE_ASYNC_LDS 1
  typedef __attribute__((address_space(1))) v4i* as1_v4i;
  typedef __attribute__((address_space(3))) v4i* as3_v4i;
  #define ASYNC_LDS_B128(gsrc, ldst) \
      __builtin_amdgcn_global_load_async_to_lds_b128((as1_v4i)(gsrc), (as3_v4i)(ldst), 0, 0)
  #if __has_builtin(__builtin_amdgcn_s_wait_asynccnt)
    #define WAIT_ASYNC(n) __builtin_amdgcn_s_wait_asynccnt(n)
  #else
    #define WAIT_ASYNC(n) asm volatile("s_wait_asynccnt " #n ::: "memory")
  #endif
#else
  #define HAVE_ASYNC_LDS 0
  #define WAIT_ASYNC(n)
#endif

__device__ inline v16bf ldfrag(const __bf16* p) {
    union { u32x4 u[2]; v16bf v; } x;
    x.u[0] = *(const u32x4*)(p);
    x.u[1] = *(const u32x4*)(p + 8);
    return x.v;
}

__device__ inline v16bf ldfrag_lds(const char* p) {
    union { u32x4 u[2]; v16bf v; } x;
    x.u[0] = *(const u32x4*)(p);
    x.u[1] = *(const u32x4*)(p + 16);
    return x.v;
}

// ---- prep: x (fp32 row-major) -> bf16 A-fragment-tiled -----------------------
__global__ __launch_bounds__(256) void ssm_conv_x(const float* __restrict__ x,
                                                  __bf16* __restrict__ xt) {
    int id = blockIdx.x * 256 + threadIdx.x;        // RT*KT*32 = 2,097,152
    int L = id & 31;
    int tile = id >> 5;
    int kt = tile & (KT - 1);
    int rt = tile >> 6;
    int half = L >> 4;
    const float* src = x + (size_t)(rt * 16 + (L & 15)) * H + kt * 32;
    __bf16* dst = xt + (size_t)id * 16;
    int k1 = 16 * half, k2 = 16 * half + 8;
#pragma unroll
    for (int i = 0; i < 8; ++i) dst[i] = (__bf16)src[k1 + i];
#pragma unroll
    for (int i = 0; i < 8; ++i) dst[8 + i] = (__bf16)src[k2 + i];
}

// ---- prep: b_mat (fp32 [h][k]) -> bf16 B-fragment-tiled ----------------------
__global__ __launch_bounds__(256) void ssm_conv_b(const float* __restrict__ b,
                                                  __bf16* __restrict__ bt) {
    int id = blockIdx.x * 256 + threadIdx.x;        // KT*NT*32 = 262,144
    int L = id & 31;
    int tile = id >> 5;
    int nt = tile & (NT - 1);
    int kt = tile >> 7;
    int half = L >> 4;
    int col = nt * 16 + (L & 15);
    __bf16* dst = bt + (size_t)id * 16;
#pragma unroll
    for (int e = 0; e < 16; ++e)
        dst[e] = (__bf16)b[(size_t)(kt * 32 + 16 * half + e) * H + col];
}

// ---- prep: a_mat -> aT bf16 B-fragment-tiled (B[j,k] = a[k,j]) ---------------
__global__ __launch_bounds__(256) void ssm_conv_a(const float* __restrict__ a,
                                                  __bf16* __restrict__ at) {
    int id = blockIdx.x * 256 + threadIdx.x;        // KT*NT*32 = 262,144
    int L = id & 31;
    int tile = id >> 5;
    int nt = tile & (NT - 1);
    int kt = tile >> 7;
    int half = L >> 4;
    const float* src = a + (size_t)(nt * 16 + (L & 15)) * H + kt * 32 + 16 * half;
    __bf16* dst = at + (size_t)id * 16;
#pragma unroll
    for (int e = 0; e < 16; ++e) dst[e] = (__bf16)src[e];
}

// ---- init: zero h double-buffer + barrier counter ----------------------------
__global__ __launch_bounds__(256) void ssm_init(unsigned int* __restrict__ hwords,
                                                unsigned int* __restrict__ cnt) {
    int id = blockIdx.x * 256 + threadIdx.x;
    if (id < HBUF) hwords[id] = 0u;                 // 32768 dwords = 128KB
    if (id == 0) *cnt = 0u;
}

// ---- phase 1: bx = x @ b_mat, 4x4 blocked WMMA, LDS-staged A -----------------
// Block = 8 waves sharing rg (same 4 A row-tiles per kt); A staged in LDS with
// double-buffered async-to-LDS loads; B streamed from L2 (per-wave tiles).
#define A_LANE_STRIDE 48                  // padded: 12 dwords -> conflict-free
#define A_TILE_BYTES  (32 * A_LANE_STRIDE)      // 1536
#define A_BUF_BYTES   (4 * A_TILE_BYTES)        // 6144 per buffer

__global__ __launch_bounds__(256) void ssm_gemm_bx(const __bf16* __restrict__ xt,
                                                   const __bf16* __restrict__ bt,
                                                   float* __restrict__ out) {
    __shared__ __attribute__((aligned(16))) char lds_a[2 * A_BUF_BYTES]; // 12KB

    const int wave = threadIdx.x >> 5;
    const int L = threadIdx.x & 31;
    const int gid = blockIdx.x * 8 + wave;          // 0..8191
    const int ng = gid & 31;                        // 32 groups of 4 N-tiles
    const int rg = gid >> 5;                        // 256 groups of 4 row-tiles

    // staging assignment: 256 threads cover 4 tiles x 64 chunks of 16B
    const int s_tile = threadIdx.x >> 6;            // 0..3
    const int s_c = threadIdx.x & 63;               // 0..63
    const int s_lds = s_tile * A_TILE_BYTES + (s_c >> 1) * A_LANE_STRIDE + (s_c & 1) * 16;
    const __bf16* s_gbase = xt + ((size_t)(rg * 4 + s_tile) * KT) * 512 + (size_t)s_c * 8;

    v8f acc[4][4] = {};

    // prologue: stage kt=0 into buffer 0
    {
        const __bf16* src = s_gbase;                // + kt*512, kt=0
        char* dst = lds_a + s_lds;
#if HAVE_ASYNC_LDS
        ASYNC_LDS_B128(src, dst);
#else
        *(u32x4*)dst = *(const u32x4*)src;
#endif
    }

    for (int kt = 0; kt < KT; ++kt) {
        if (kt + 1 < KT) {                          // stage kt+1 into other buffer
            const __bf16* src = s_gbase + (size_t)(kt + 1) * 512;
            char* dst = lds_a + ((kt + 1) & 1) * A_BUF_BYTES + s_lds;
#if HAVE_ASYNC_LDS
            ASYNC_LDS_B128(src, dst);
#else
            *(u32x4*)dst = *(const u32x4*)src;
#endif
            WAIT_ASYNC(1);                          // current buffer's DMA done
        } else {
            WAIT_ASYNC(0);
        }
        __syncthreads();                            // all chunks of buf[kt&1] visible

        const char* abase = lds_a + (kt & 1) * A_BUF_BYTES + L * A_LANE_STRIDE;
        v16bf af[4], bf[4];
#pragma unroll
        for (int i = 0; i < 4; ++i)
            af[i] = ldfrag_lds(abase + i * A_TILE_BYTES);
#pragma unroll
        for (int j = 0; j < 4; ++j)
            bf[j] = ldfrag(bt + ((size_t)(kt * NT + ng * 4 + j) * 32 + L) * 16);
#pragma unroll
        for (int i = 0; i < 4; ++i)
#pragma unroll
            for (int j = 0; j < 4; ++j)
                acc[i][j] = __builtin_amdgcn_wmma_f32_16x16x32_bf16(
                    false, af[i], false, bf[j], (short)0, acc[i][j], false, false);

        __syncthreads();                            // readers done before overwrite
    }

    const int n = L & 15, half = L >> 4;
#pragma unroll
    for (int i = 0; i < 4; ++i)
#pragma unroll
        for (int j = 0; j < 4; ++j) {
            const int row0 = (rg * 4 + i) * 16 + 8 * half;
            const int col = (ng * 4 + j) * 16 + n;
#pragma unroll
            for (int r = 0; r < 8; ++r)
                out[(size_t)(row0 + r) * H + col] = acc[i][j][r];
        }
}

// ---- phase 2: persistent scan, 8 WGs x 8 waves, 2 N-tiles per wave -----------
__global__ __launch_bounds__(256) void ssm_scan(const __bf16* __restrict__ at,
                                                float* __restrict__ io,
                                                const unsigned char* __restrict__ reset,
                                                __bf16* __restrict__ h,
                                                unsigned int* __restrict__ cnt) {
    const int wave = threadIdx.x >> 5;
    const int L = threadIdx.x & 31;
    const int nt0 = (blockIdx.x * 8 + wave) * 2;    // 0,2,..,126
    const int n = L & 15, half = L >> 4;
    const int col0 = nt0 * 16;
    const int hb = L & 15;                          // A-frag row (batch) for this lane
    const v16bf zero = {};

#pragma unroll 1
    for (int t = 0; t < T; ++t) {
        const __bf16* hcur = h + (size_t)(t & 1) * HBUF;
        __bf16* hnxt = h + (size_t)((t + 1) & 1) * HBUF;
        const bool rz = (hb < BB) && (reset[hb * T + t] != 0);

        // C = bx tiles (rows 8..15 are padding -> 0)
        v8f acc0, acc1;
#pragma unroll
        for (int r = 0; r < 8; ++r) {
            int b = r + 8 * half;
            float c0 = 0.0f, c1 = 0.0f;
            if (b < BB) {
                const float* p = io + ((size_t)b * T + t) * H + col0 + n;
                c0 = p[0];
                c1 = p[16];
            }
            acc0[r] = c0;
            acc1[r] = c1;
        }

#pragma unroll 4
        for (int kt = 0; kt < KT; ++kt) {
            // A-frag directly from row-major bf16 h (L2/WGP$-hot): two b128 runs
            union { u32x4 u[2]; v16bf v; } af;
            const __bf16* hp = hcur + hb * H + kt * 32 + 16 * half;
            af.u[0] = *(const u32x4*)(hp);
            af.u[1] = *(const u32x4*)(hp + 8);
            v16bf a = af.v;
            if (rz) a = zero;                       // per-lane reset of h row

            const __bf16* bp = at + ((size_t)(kt * NT + nt0) * 32 + L) * 16;
            __builtin_prefetch(bp + (size_t)32 * NT * 16, 0, 1);   // next kt
            v16bf bf0 = ldfrag(bp);
            v16bf bf1 = ldfrag(bp + 512);

            acc0 = __builtin_amdgcn_wmma_f32_16x16x32_bf16(
                false, a, false, bf0, (short)0, acc0, false, false);
            acc1 = __builtin_amdgcn_wmma_f32_16x16x32_bf16(
                false, a, false, bf1, (short)0, acc1, false, false);
        }

        // write h_t to output (fp32) and to next bf16 buffer
#pragma unroll
        for (int r = 0; r < 8; ++r) {
            int b = r + 8 * half;
            if (b < BB) {
                float v0 = acc0[r], v1 = acc1[r];
                float* p = io + ((size_t)b * T + t) * H + col0 + n;
                p[0] = v0;
                p[16] = v1;
                __bf16* q = hnxt + b * H + col0 + n;
                q[0] = (__bf16)v0;
                q[16] = (__bf16)v1;
            }
        }

        // device-wide barrier between steps (monotonic counter through L2)
        __threadfence();
        __syncthreads();
        if (threadIdx.x == 0) {
            __hip_atomic_fetch_add(cnt, 1u, __ATOMIC_RELEASE, __HIP_MEMORY_SCOPE_AGENT);
            const unsigned target = (unsigned)NWG * (unsigned)(t + 1);
            while (__hip_atomic_load(cnt, __ATOMIC_ACQUIRE, __HIP_MEMORY_SCOPE_AGENT) < target)
                __builtin_amdgcn_s_sleep(2);
        }
        __syncthreads();
        __threadfence();                            // invalidate WGP$ before next-step h reads
    }
}

// ---------------------------------------------------------------------------
extern "C" void kernel_launch(void* const* d_in, const int* in_sizes, int n_in,
                              void* d_out, int out_size, void* d_ws, size_t ws_size,
                              hipStream_t stream) {
    const float* x = (const float*)d_in[0];
    const unsigned char* reset = (const unsigned char*)d_in[1];
    const float* a_mat = (const float*)d_in[2];
    const float* b_mat = (const float*)d_in[3];
    float* out = (float*)d_out;

    // Workspace layout (needs ~84.2 MB):
    //   [0,64MB)      x bf16, A-frag tiled
    //   [64,72MB)     b_mat bf16, B-frag tiled
    //   [72,80MB)     a^T bf16, B-frag tiled
    //   [80MB,+128KB) h double buffer (bf16, row-major 16x2048 x2)
    //   [+256KB]      barrier counter
    char* ws = (char*)d_ws;
    const size_t XT = 67108864, MT = 8388608;
    __bf16* xt = (__bf16*)(ws);
    __bf16* bt = (__bf16*)(ws + XT);
    __bf16* at = (__bf16*)(ws + XT + MT);
    __bf16* h  = (__bf16*)(ws + XT + 2 * MT);
    unsigned int* cnt = (unsigned int*)(ws + XT + 2 * MT + 262144);

    ssm_conv_x<<<8192, 256, 0, stream>>>(x, xt);
    ssm_conv_b<<<1024, 256, 0, stream>>>(b_mat, bt);
    ssm_conv_a<<<1024, 256, 0, stream>>>(a_mat, at);
    ssm_init<<<128, 256, 0, stream>>>((unsigned int*)h, cnt);
    ssm_gemm_bx<<<1024, 256, 0, stream>>>(xt, bt, out);
    ssm_scan<<<NWG, 256, 0, stream>>>(at, out, reset, h, cnt);
}
